// RegionProposalNetwork_4698694222526
// MI455X (gfx1250) — matrix operand order repaired and checked by reading, not compiled, not run on hardware
//
#include <hip/hip_runtime.h>
#include <hip/hip_bf16.h>

typedef unsigned short u16;
typedef unsigned int u32;
typedef __bf16 bf16x16 __attribute__((ext_vector_type(16)));
typedef float  f32x8   __attribute__((ext_vector_type(8)));
typedef int    v4i_g   __attribute__((__vector_size__(16)));   // matches builtin pointee

union Frag { uint4 q[2]; bf16x16 m; };

// ---------------- constants ----------------
#define N_TOT   774144      // total anchors
#define AB1     589824      // anchor base level1
#define AB2     737280      // anchor base level2
#define LOCS_OFF  0
#define SC_OFF    3096576   // 774144*4
#define ROIS_OFF  4644864   // SC_OFF + 774144*2
#define RFG_OFF   4652864
#define RIDX_OFF  4654864
#define ANCH_OFF  4656864
#define N_PRE   12000
#define N_POST  2000
#define N_CAND  16384
#define MASK_W  376         // 12000/32 = 375 words used, +1 spare

// ---------------- async global->LDS staging (CDNA5) ----------------
#if defined(__HIP_DEVICE_COMPILE__) && __has_builtin(__builtin_amdgcn_global_load_async_to_lds_b128)
#define HAS_ASYNC_LDS 1
__device__ __forceinline__ void cp16(uint4* l, const uint4* g) {
  v4i_g* gp = reinterpret_cast<v4i_g*>(const_cast<uint4*>(g));
  v4i_g* lp = reinterpret_cast<v4i_g*>(l);
  __builtin_amdgcn_global_load_async_to_lds_b128(
      (__attribute__((address_space(1))) v4i_g*)gp,
      (__attribute__((address_space(3))) v4i_g*)lp, 0, 0);
}
__device__ __forceinline__ void cp_wait() {
#if __has_builtin(__builtin_amdgcn_s_wait_asynccnt)
  __builtin_amdgcn_s_wait_asynccnt(0);
#else
  asm volatile("s_wait_asynccnt 0x0" ::: "memory");
#endif
}
#else
__device__ __forceinline__ void cp16(uint4* l, const uint4* g) { *l = *g; }
__device__ __forceinline__ void cp_wait() {}
#endif

__device__ __forceinline__ u16 f2bf(float f) {
  u32 u = __float_as_uint(f);
  u32 r = u + 0x7FFFu + ((u >> 16) & 1u);
  return (u16)(r >> 16);
}
__device__ __forceinline__ u32 flipf(float f) {
  u32 u = __float_as_uint(f);
  return (u >> 31) ? ~u : (u | 0x80000000u);
}
__device__ __forceinline__ float unflipf(u32 k) {
  u32 u = (k >> 31) ? (k ^ 0x80000000u) : ~k;
  return __uint_as_float(u);
}

// ---------------- weight packing ----------------
// conv1_w (co,ci,3,3) f32 -> Wp[tap][co][ci] bf16
__global__ void pack_conv_w(const float* __restrict__ w, u16* __restrict__ wp) {
  int idx = blockIdx.x * 256 + threadIdx.x;           // 9*256*256
  if (idx >= 9 * 256 * 256) return;
  int ci = idx & 255, co = (idx >> 8) & 255, tap = idx >> 16;
  int ky = tap / 3, kx = tap % 3;
  wp[idx] = f2bf(w[((co * 256 + ci) * 3 + ky) * 3 + kx]);
}
// head weights -> Whead[n][ci] bf16 : n<18 score, 18<=n<54 loc, else 0
__global__ void pack_head_w(const float* __restrict__ sw, const float* __restrict__ lw,
                            u16* __restrict__ wh) {
  int idx = blockIdx.x * 256 + threadIdx.x;           // 64*256
  if (idx >= 64 * 256) return;
  int ci = idx & 255, n = idx >> 8;
  float v = 0.f;
  if (n < 18) v = sw[n * 256 + ci];
  else if (n < 54) v = lw[(n - 18) * 256 + ci];
  wh[idx] = f2bf(v);
}

// ---------------- NCHW f32 -> padded NHWC bf16 ----------------
__global__ void pad_convert(const float* __restrict__ src, u16* __restrict__ dst,
                            int H, int W) {
  int idx = blockIdx.x * 256 + threadIdx.x;
  int total = (H + 2) * (W + 2) * 256;
  if (idx >= total) return;
  int c = idx & 255, pix = idx >> 8;
  int W2 = W + 2;
  int px = pix % W2, py = pix / W2;
  float v = 0.f;
  if (py >= 1 && py <= H && px >= 1 && px <= W)
    v = src[(c * H + (py - 1)) * W + (px - 1)];
  dst[idx] = f2bf(v);
}

// ---------------- 3x3 conv as implicit GEMM, bf16 WMMA ----------------
// block tile: 64 (M=spatial) x 128 (N=out ch); 8 waves of 32x32.
// Double-buffered LDS; async global->LDS copies hide latency behind WMMAs.
__global__ __launch_bounds__(256) void conv3x3_wmma(
    const u16* __restrict__ pad, const u16* __restrict__ wp,
    const float* __restrict__ bias, u16* __restrict__ out, int W) {
  const int W2 = W + 2;
  __shared__ alignas(16) u16 As[2 * 64 * 40];    // 80B row stride, 2 buffers
  __shared__ alignas(16) u16 Bs[2 * 128 * 40];
  uint4* As4 = (uint4*)As;   // buffer stride 320 uint4
  uint4* Bs4 = (uint4*)Bs;   // buffer stride 640 uint4

  const int tid = threadIdx.x;
  const int p0 = blockIdx.x * 64;
  const int nblk = blockIdx.y * 128;

  const int arow = tid >> 2, aseg = tid & 3;       // A: 64 rows x 4x16B
  const int p = p0 + arow;
  const int ay = p / W, ax = p % W;
  const int brow = tid >> 1, bseg = (tid & 1) * 2; // B: 128 rows x 2x16B/thread

  const int lane = tid & 31, half = lane >> 4;
  const int wv = tid >> 5;
  const int m0 = (wv & 1) * 32, n0 = (wv >> 1) * 32;

  auto issue = [&](int it, int buf) {
    const int tap = it >> 3, ci0 = (it & 7) * 32;
    const int ky = tap / 3, kx = tap - ky * 3;
    const long aoff = ((long)(ay + ky) * W2 + (ax + kx)) * 256;
    const long boff = ((long)tap * 256 + (nblk + brow)) * 256;
    cp16(&As4[buf * 320 + arow * 5 + aseg], (const uint4*)(pad + aoff + ci0) + aseg);
    const uint4* gb = (const uint4*)(wp + boff + ci0);
    cp16(&Bs4[buf * 640 + brow * 5 + bseg],     gb + bseg);
    cp16(&Bs4[buf * 640 + brow * 5 + bseg + 1], gb + bseg + 1);
  };

  const f32x8 vzero = {0.f,0.f,0.f,0.f,0.f,0.f,0.f,0.f};
  f32x8 acc[2][2];
  acc[0][0] = vzero; acc[0][1] = vzero; acc[1][0] = vzero; acc[1][1] = vzero;

  issue(0, 0);
  #pragma unroll 1
  for (int it = 0; it < 72; ++it) {
    const int buf = it & 1;
    cp_wait();            // my async copies for this buffer are done
    __syncthreads();      // everyone's copies are done
    if (it + 1 < 72) issue(it + 1, buf ^ 1);   // prefetch next K-slice
    Frag aF[2], bF[2];
    #pragma unroll
    for (int ms = 0; ms < 2; ++ms) {
      int r = buf * 320 + (m0 + ms * 16 + (lane & 15)) * 5;
      aF[ms].q[0] = As4[r + half];
      aF[ms].q[1] = As4[r + 2 + half];
    }
    #pragma unroll
    for (int ns = 0; ns < 2; ++ns) {
      int r = buf * 640 + (n0 + ns * 16 + (lane & 15)) * 5;
      bF[ns].q[0] = Bs4[r + half];
      bF[ns].q[1] = Bs4[r + 2 + half];
    }
    #pragma unroll
    for (int ms = 0; ms < 2; ++ms)
      #pragma unroll
      for (int ns = 0; ns < 2; ++ns)
        acc[ms][ns] = __builtin_amdgcn_wmma_f32_16x16x32_bf16(
            false, aF[ms].m, false, bF[ns].m, (short)0, acc[ms][ns], false, false);
  }
  #pragma unroll
  for (int ms = 0; ms < 2; ++ms)
    #pragma unroll
    for (int ns = 0; ns < 2; ++ns) {
      const int n = nblk + n0 + ns * 16 + (lane & 15);
      const float b = bias[n];
      #pragma unroll
      for (int j = 0; j < 8; ++j) {
        const int m = m0 + ms * 16 + j + half * 8;
        float v = acc[ms][ns][j] + b;
        v = v > 0.f ? v : 0.f;
        out[(long)(p0 + m) * 256 + n] = f2bf(v);
      }
    }
}

// ---------------- fused 1x1 heads: [HW,256] x [256,64] WMMA ----------------
// block tile: 128 (M) x 64 (N); 8 waves = 4(M) x 2(N); double-buffered LDS
__global__ __launch_bounds__(256) void head_wmma(
    const u16* __restrict__ inp, const u16* __restrict__ wh,
    const float* __restrict__ score_b, const float* __restrict__ loc_b,
    float* __restrict__ locs_out, float* __restrict__ sc_out, long abase) {
  __shared__ alignas(16) u16 As[2 * 128 * 40];
  __shared__ alignas(16) u16 Bs[2 * 64 * 40];
  uint4* As4 = (uint4*)As;   // buffer stride 640
  uint4* Bs4 = (uint4*)Bs;   // buffer stride 320

  const int tid = threadIdx.x;
  const long p0 = (long)blockIdx.x * 128;
  const int arow = tid >> 1, aseg = (tid & 1) * 2;
  const int brow = tid >> 2, bseg = tid & 3;
  const int lane = tid & 31, half = lane >> 4;
  const int wv = tid >> 5;
  const int m0 = (wv & 3) * 32, n0 = (wv >> 2) * 32;

  auto issue = [&](int kc, int buf) {
    const int ci0 = kc * 32;
    const uint4* ga = (const uint4*)(inp + (p0 + arow) * 256 + ci0);
    cp16(&As4[buf * 640 + arow * 5 + aseg],     ga + aseg);
    cp16(&As4[buf * 640 + arow * 5 + aseg + 1], ga + aseg + 1);
    const uint4* gb = (const uint4*)(wh + (long)brow * 256 + ci0);
    cp16(&Bs4[buf * 320 + brow * 5 + bseg], gb + bseg);
  };

  const f32x8 vzero = {0.f,0.f,0.f,0.f,0.f,0.f,0.f,0.f};
  f32x8 acc[2][2];
  acc[0][0] = vzero; acc[0][1] = vzero; acc[1][0] = vzero; acc[1][1] = vzero;

  issue(0, 0);
  #pragma unroll 1
  for (int kc = 0; kc < 8; ++kc) {
    const int buf = kc & 1;
    cp_wait();
    __syncthreads();
    if (kc + 1 < 8) issue(kc + 1, buf ^ 1);
    Frag aF[2], bF[2];
    #pragma unroll
    for (int ms = 0; ms < 2; ++ms) {
      int r = buf * 640 + (m0 + ms * 16 + (lane & 15)) * 5;
      aF[ms].q[0] = As4[r + half];
      aF[ms].q[1] = As4[r + 2 + half];
    }
    #pragma unroll
    for (int ns = 0; ns < 2; ++ns) {
      int r = buf * 320 + (n0 + ns * 16 + (lane & 15)) * 5;
      bF[ns].q[0] = Bs4[r + half];
      bF[ns].q[1] = Bs4[r + 2 + half];
    }
    #pragma unroll
    for (int ms = 0; ms < 2; ++ms)
      #pragma unroll
      for (int ns = 0; ns < 2; ++ns)
        acc[ms][ns] = __builtin_amdgcn_wmma_f32_16x16x32_bf16(
            false, aF[ms].m, false, bF[ns].m, (short)0, acc[ms][ns], false, false);
  }
  #pragma unroll
  for (int ms = 0; ms < 2; ++ms)
    #pragma unroll
    for (int ns = 0; ns < 2; ++ns) {
      const int n = n0 + ns * 16 + (lane & 15);
      float bb = 0.f;
      if (n < 18) bb = score_b[n];
      else if (n < 54) bb = loc_b[n - 18];
      #pragma unroll
      for (int j = 0; j < 8; ++j) {
        const int m = m0 + ms * 16 + j + half * 8;
        const long p = p0 + m;
        const float v = acc[ms][ns][j] + bb;
        if (n < 18) {
          long a = n >> 1; int c = n & 1;
          sc_out[((abase + p * 9 + a) << 1) + c] = v;
        } else if (n < 54) {
          int mm = n - 18; long a = mm >> 2; int dd = mm & 3;
          locs_out[((abase + p * 9 + a) << 2) + dd] = v;
        }
      }
    }
}

// ---------------- fg softmax (index 1) ----------------
__global__ void fg_kernel(const float* __restrict__ sc, float* __restrict__ fg) {
  int g = blockIdx.x * 256 + threadIdx.x;
  if (g >= N_TOT) return;
  float s0 = sc[2 * g], s1 = sc[2 * g + 1];
  fg[g] = 1.f / (1.f + __expf(s0 - s1));
}

// ---------------- anchors ----------------
__global__ void anchors_kernel(float* __restrict__ anch) {
  int g = blockIdx.x * 256 + threadIdx.x;
  if (g >= N_TOT) return;
  int base, W, stride;
  if (g < AB1)      { base = g;       W = 256; stride = 4;  }
  else if (g < AB2) { base = g - AB1; W = 128; stride = 8;  }
  else              { base = g - AB2; W = 64;  stride = 16; }
  int p = base / 9, a = base % 9;
  int y = p / W, x = p % W;
  const float ratios[3] = {0.5f, 1.f, 2.f};
  const float sizes[3]  = {8.f, 16.f, 32.f};
  float r = ratios[a / 3], s = sizes[a % 3];
  float h = s * __fsqrt_rn(r), w = s * __fsqrt_rn(1.f / r);
  float ys = (float)(y * stride), xs = (float)(x * stride);
  anch[4 * g + 0] = ys - 0.5f * h;
  anch[4 * g + 1] = xs - 0.5f * w;
  anch[4 * g + 2] = ys + 0.5f * h;
  anch[4 * g + 3] = xs + 0.5f * w;
}

// ---------------- decode + clip + filter ----------------
__global__ void proposals_kernel(const float* __restrict__ anch, const float* __restrict__ loc,
                                 const float* __restrict__ fg,
                                 float4* __restrict__ boxes, u32* __restrict__ key) {
  int g = blockIdx.x * 256 + threadIdx.x;
  if (g >= N_TOT) return;
  float a0 = anch[4*g], a1 = anch[4*g+1], a2 = anch[4*g+2], a3 = anch[4*g+3];
  float ah = a2 - a0, aw = a3 - a1;
  float acy = a0 + 0.5f * ah, acx = a1 + 0.5f * aw;
  float dy = loc[4*g], dx = loc[4*g+1], dh = loc[4*g+2], dw = loc[4*g+3];
  float cy = dy * ah + acy, cx = dx * aw + acx;
  float h = __expf(dh) * ah, w = __expf(dw) * aw;
  float y1 = fminf(fmaxf(cy - 0.5f * h, 0.f), 1024.f);
  float x1 = fminf(fmaxf(cx - 0.5f * w, 0.f), 1024.f);
  float y2 = fminf(fmaxf(cy + 0.5f * h, 0.f), 1024.f);
  float x2 = fminf(fmaxf(cx + 0.5f * w, 0.f), 1024.f);
  boxes[g] = make_float4(y1, x1, y2, x2);
  bool valid = (y2 - y1 >= 16.f) && (x2 - x1 >= 16.f);
  float s = valid ? fg[g] : __uint_as_float(0xFF800000u);  // -inf
  key[g] = flipf(s);
}

// ---------------- top-k selection ----------------
__global__ void hist_zero(u32* __restrict__ hist) {
  int i = blockIdx.x * 256 + threadIdx.x;
  if (i < 4096 + 64) hist[i] = 0;                  // hist + ctrl block
}
__global__ void hist_kernel(const u32* __restrict__ key, u32* __restrict__ hist) {
  int g = blockIdx.x * 256 + threadIdx.x;
  if (g >= N_TOT) return;
  atomicAdd(&hist[key[g] >> 20], 1u);
}
__global__ void pick_threshold(const u32* __restrict__ hist, u32* __restrict__ ctrl) {
  if (threadIdx.x || blockIdx.x) return;
  u32 c = 0; int b = 4095;
  for (; b >= 0; --b) { c += hist[b]; if (c >= (u32)N_PRE) break; }
  if (b < 0) b = 0;
  ctrl[0] = (u32)b << 20;
}
__global__ void cand_init(u32* __restrict__ ck, u32* __restrict__ ci) {
  int i = blockIdx.x * 256 + threadIdx.x;
  if (i < N_CAND) { ck[i] = 0u; ci[i] = 0u; }
}
__global__ void compact_kernel(const u32* __restrict__ key, u32* __restrict__ ctrl,
                               u32* __restrict__ ck, u32* __restrict__ ci) {
  int g = blockIdx.x * 256 + threadIdx.x;
  if (g >= N_TOT) return;
  u32 k = key[g];
  if (k >= ctrl[0]) {
    u32 pos = atomicAdd(&ctrl[1], 1u);
    if (pos < N_CAND) { ck[pos] = k; ci[pos] = (u32)g; }
  }
}
__global__ void bitonic_step(u32* __restrict__ key, u32* __restrict__ idx, int k, int j) {
  int i = blockIdx.x * 256 + threadIdx.x;
  if (i >= N_CAND) return;
  int l = i ^ j;
  if (l > i) {
    u32 ki = key[i], kl = key[l];
    bool up = ((i & k) == 0);                      // descending overall
    if (up ? (ki < kl) : (ki > kl)) {
      key[i] = kl; key[l] = ki;
      u32 t = idx[i]; idx[i] = idx[l]; idx[l] = t;
    }
  }
}
__global__ void gather_top(const u32* __restrict__ ck, const u32* __restrict__ ci,
                           const float4* __restrict__ boxes,
                           float4* __restrict__ tbox, float* __restrict__ tscore,
                           int* __restrict__ tvalid) {
  int i = blockIdx.x * 256 + threadIdx.x;
  if (i >= N_PRE) return;
  u32 k = ck[i], g = ci[i];
  float s = unflipf(k);
  tbox[i] = boxes[g];
  tscore[i] = s;
  tvalid[i] = (k != 0u) && __builtin_isfinite(s);
}

// ---------------- NMS ----------------
__global__ void nms_mask(const float4* __restrict__ tbox, u32* __restrict__ mask) {
  int i = blockIdx.x;
  float4 bi = tbox[i];
  float ai = (bi.z - bi.x) * (bi.w - bi.y);
  for (int w = threadIdx.x; w < MASK_W; w += blockDim.x) {
    u32 bits = 0u;
    for (int b = 0; b < 32; ++b) {
      int j = w * 32 + b;
      if (j > i && j < N_PRE) {
        float4 bj = tbox[j];
        float yy1 = fmaxf(bi.x, bj.x), xx1 = fmaxf(bi.y, bj.y);
        float yy2 = fminf(bi.z, bj.z), xx2 = fminf(bi.w, bj.w);
        float inter = fmaxf(yy2 - yy1, 0.f) * fmaxf(xx2 - xx1, 0.f);
        float aj = (bj.z - bj.x) * (bj.w - bj.y);
        float iou = inter / (ai + aj - inter + 1e-9f);
        if (iou > 0.7f) bits |= (1u << b);
      }
    }
    mask[(long)i * MASK_W + w] = bits;
  }
}
__global__ void nms_scan(const u32* __restrict__ mask, const int* __restrict__ tvalid,
                         int* __restrict__ keep) {
  __shared__ u32 removed[MASK_W];
  __shared__ int keepi;
  for (int w = threadIdx.x; w < MASK_W; w += blockDim.x) removed[w] = 0u;
  __syncthreads();
  for (int i = 0; i < N_PRE; ++i) {
    if (threadIdx.x == 0) {
      bool r = (removed[i >> 5] >> (i & 31)) & 1u;
      keepi = (!r) && tvalid[i];
      keep[i] = keepi;
    }
    __syncthreads();
    if (keepi)
      for (int w = threadIdx.x; w < MASK_W; w += blockDim.x)
        removed[w] |= mask[(long)i * MASK_W + w];
    __syncthreads();
  }
}
__global__ void finalize_kernel(const int* __restrict__ keep, const float4* __restrict__ tbox,
                                const float* __restrict__ tscore, float* __restrict__ out) {
  if (threadIdx.x || blockIdx.x) return;
  float* rois = out + ROIS_OFF;
  float* rfg  = out + RFG_OFF;
  float* ridx = out + RIDX_OFF;
  int cnt = 0;
  for (int i = 0; i < N_PRE && cnt < N_POST; ++i)
    if (keep[i]) {
      float4 b = tbox[i];
      rois[cnt*4+0] = b.x; rois[cnt*4+1] = b.y; rois[cnt*4+2] = b.z; rois[cnt*4+3] = b.w;
      rfg[cnt] = tscore[i];
      ++cnt;
    }
  for (; cnt < N_POST; ++cnt) {
    rois[cnt*4+0] = rois[cnt*4+1] = rois[cnt*4+2] = rois[cnt*4+3] = 0.f;
    rfg[cnt] = 0.f;
  }
  for (int i = 0; i < N_POST; ++i) ridx[i] = 0.f;
}

// ---------------- launcher ----------------
static constexpr size_t au(size_t x) { return (x + 255) & ~size_t(255); }

extern "C" void kernel_launch(void* const* d_in, const int* in_sizes, int n_in,
                              void* d_out, int out_size, void* d_ws, size_t ws_size,
                              hipStream_t stream) {
  const float* ft[3]    = {(const float*)d_in[0], (const float*)d_in[1], (const float*)d_in[2]};
  const float* conv1_w  = (const float*)d_in[3];
  const float* conv1_b  = (const float*)d_in[4];
  const float* score_w  = (const float*)d_in[5];
  const float* score_b  = (const float*)d_in[6];
  const float* loc_w    = (const float*)d_in[7];
  const float* loc_b    = (const float*)d_in[8];
  float* out = (float*)d_out;
  char*  ws  = (char*)d_ws;

  const int   Hs[3]  = {256, 128, 64};
  const long  HWs[3] = {65536, 16384, 4096};
  const long  abase[3] = {0, AB1, AB2};

  size_t off = 0;
  u16* pad[3];
  pad[0] = (u16*)(ws + off); off += au((size_t)258*258*256*2);
  pad[1] = (u16*)(ws + off); off += au((size_t)130*130*256*2);
  pad[2] = (u16*)(ws + off); off += au((size_t)66*66*256*2);
  u16* wp    = (u16*)(ws + off); off += au((size_t)9*256*256*2);
  u16* wh    = (u16*)(ws + off); off += au((size_t)64*256*2);
  u16* co[3];
  co[0] = (u16*)(ws + off); off += au((size_t)65536*256*2);
  co[1] = (u16*)(ws + off); off += au((size_t)16384*256*2);
  co[2] = (u16*)(ws + off); off += au((size_t)4096*256*2);
  float*  fg    = (float*)(ws + off);  off += au((size_t)N_TOT*4);
  float4* boxes = (float4*)(ws + off); off += au((size_t)N_TOT*16);
  u32*    key   = (u32*)(ws + off);    off += au((size_t)N_TOT*4);
  u32*    hist  = (u32*)(ws + off);    off += au((size_t)4096*4 + 64*4);
  u32*    ctrl  = hist + 4096;
  u32*    ck    = (u32*)(ws + off);    off += au((size_t)N_CAND*4);
  u32*    ci    = (u32*)(ws + off);    off += au((size_t)N_CAND*4);
  float4* tbox  = (float4*)(ws + off); off += au((size_t)N_PRE*16);
  float*  tscore= (float*)(ws + off);  off += au((size_t)N_PRE*4);
  int*    tvalid= (int*)(ws + off);    off += au((size_t)N_PRE*4);
  u32*    mask  = (u32*)(ws + off);    off += au((size_t)N_PRE*MASK_W*4);
  int*    keep  = (int*)(ws + off);    off += au((size_t)N_PRE*4);

  // 1) weight packing
  pack_conv_w<<<(9*256*256)/256, 256, 0, stream>>>(conv1_w, wp);
  pack_head_w<<<(64*256)/256, 256, 0, stream>>>(score_w, loc_w, wh);

  // 2) pad + convert inputs
  for (int L = 0; L < 3; ++L) {
    int H = Hs[L], W = Hs[L];
    int total = (H + 2) * (W + 2) * 256;
    pad_convert<<<(total + 255)/256, 256, 0, stream>>>(ft[L], pad[L], H, W);
  }

  // 3) 3x3 conv + ReLU (bf16 WMMA implicit GEMM, async double-buffered)
  for (int L = 0; L < 3; ++L) {
    dim3 grid((unsigned)(HWs[L] / 64), 2);
    conv3x3_wmma<<<grid, 256, 0, stream>>>(pad[L], wp, conv1_b, co[L], Hs[L]);
  }

  // 4) fused 1x1 heads (score+loc) WMMA, writes rpn_locs / rpn_scores
  for (int L = 0; L < 3; ++L) {
    head_wmma<<<(unsigned)(HWs[L] / 128), 256, 0, stream>>>(
        co[L], wh, score_b, loc_b, out + LOCS_OFF, out + SC_OFF, abase[L]);
  }

  // 5) fg softmax, anchors, proposals
  const int gN = (N_TOT + 255) / 256;
  fg_kernel<<<gN, 256, 0, stream>>>(out + SC_OFF, fg);
  anchors_kernel<<<gN, 256, 0, stream>>>(out + ANCH_OFF);
  proposals_kernel<<<gN, 256, 0, stream>>>(out + ANCH_OFF, out + LOCS_OFF, fg, boxes, key);

  // 6) top-N_PRE selection via histogram threshold + compaction + bitonic sort
  hist_zero<<<(4096 + 64 + 255)/256, 256, 0, stream>>>(hist);
  hist_kernel<<<gN, 256, 0, stream>>>(key, hist);
  pick_threshold<<<1, 32, 0, stream>>>(hist, ctrl);
  cand_init<<<N_CAND/256, 256, 0, stream>>>(ck, ci);
  compact_kernel<<<gN, 256, 0, stream>>>(key, ctrl, ck, ci);
  for (int k = 2; k <= N_CAND; k <<= 1)
    for (int j = k >> 1; j > 0; j >>= 1)
      bitonic_step<<<N_CAND/256, 256, 0, stream>>>(ck, ci, k, j);
  gather_top<<<(N_PRE + 255)/256, 256, 0, stream>>>(ck, ci, boxes, tbox, tscore, tvalid);

  // 7) NMS + finalize
  nms_mask<<<N_PRE, 256, 0, stream>>>(tbox, mask);
  nms_scan<<<1, 256, 0, stream>>>(mask, tvalid, keep);
  finalize_kernel<<<1, 32, 0, stream>>>(keep, tbox, tscore, out);
}
